// ScaledDotProductAttention_22067541967011
// MI455X (gfx1250) — matrix operand order repaired and checked by reading, not compiled, not run on hardware
//
#include <hip/hip_runtime.h>
#include <cstdint>
#include <cstddef>

// ---------------------------------------------------------------------------
// Scaled dot-product attention (with materialized attn matrix), MI455X/gfx1250
//   out   = softmax(mask(Q K^T / 32)) V      [16, 2048, 128]  fp32
//   attn  = softmax(mask(Q K^T / 32))        [16, 2048, 2048] fp32
// Strategy: 1 workgroup (8 wave32) per (batch, 32-row query tile).
//   - 32x2048 fp32 score strip resident in 320KB WGP LDS (256 KB)
//   - QK^T and PV via v_wmma_f32_16x16x32_f16 (f16 inputs, f32 accum)
//   - softmax in fp32, attn/mask streamed with non-temporal hints
// ---------------------------------------------------------------------------

typedef _Float16 v16h __attribute__((ext_vector_type(16)));
typedef _Float16 h8   __attribute__((ext_vector_type(8)));
typedef float    v8f  __attribute__((ext_vector_type(8)));

constexpr int   L       = 2048;   // Lq == Lk
constexpr int   D       = 128;    // Dk == Dv
constexpr int   MTILE   = 32;     // query rows per workgroup
constexpr float INV_TEMPER = 1.0f / 32.0f;   // 1/sqrt(1024)

// LDS layout (dynamic shared):
//   [0, 256K)          : S, 32 x 2048 fp32 scores; later overlaid by P f16
//                        (P row r lives in the first half of S row r's bytes)
//   [256K, 256K+8K)    : Qs, 32 x 128 f16 query tile
//   [256K+8K, +8K)     : Vt, 128 x 32 f16 transposed V chunk
constexpr size_t SMEM_S  = (size_t)MTILE * L * 4;
constexpr size_t SMEM_Q  = (size_t)MTILE * D * 2;
constexpr size_t SMEM_V  = (size_t)D * 32 * 2;
constexpr size_t SMEM_SZ = SMEM_S + SMEM_Q + SMEM_V;   // 278528 B < 320 KB

__device__ __forceinline__ v8f wmma_f16(v16h a, v16h b, v8f c) {
  return __builtin_amdgcn_wmma_f32_16x16x32_f16(false, a, false, b,
                                                (short)0, c, false, false);
}

__global__ __launch_bounds__(256, 1)
void sdpa_attn_kernel(const float* __restrict__ Q,
                      const float* __restrict__ K,
                      const float* __restrict__ V,
                      const uint8_t* __restrict__ MASK,
                      float* __restrict__ OUT,
                      float* __restrict__ ATTN) {
  extern __shared__ char smem[];
  float*    S  = reinterpret_cast<float*>(smem);                       // 32x2048 f32
  _Float16* Ph = reinterpret_cast<_Float16*>(smem);                    // P overlay; row r at half-index r*4096
  _Float16* Qs = reinterpret_cast<_Float16*>(smem + SMEM_S);           // 32x128 f16
  _Float16* Vt = reinterpret_cast<_Float16*>(smem + SMEM_S + SMEM_Q);  // 128x32 f16

  const int tid  = threadIdx.x;
  const int lane = tid & 31;
  const int w    = tid >> 5;          // wave id 0..7
  const int l16  = lane & 15;
  const int lhi  = lane >> 4;         // lane half (0/1)

  const int qt = blockIdx.x;          // query tile 0..63
  const int b  = blockIdx.y;          // batch 0..15
  const int q0 = qt * MTILE;

  // ------------------------------------------------------------------
  // Phase 0: stage Q tile into LDS as f16 (coalesced, 16 elems/thread)
  // ------------------------------------------------------------------
  #pragma unroll
  for (int i = 0; i < 16; ++i) {
    int e = i * 256 + tid;
    int r = e >> 7, d = e & 127;
    Qs[r * D + d] = (_Float16)Q[((size_t)b * L + q0 + r) * D + d];
  }
  __syncthreads();

  // ------------------------------------------------------------------
  // Phase 1: S = Q K^T  (2 row-blocks x 128 col-blocks of 16x16 tiles)
  //   wave w owns row-block (w&1) and column residue class (w>>1) mod 4
  // ------------------------------------------------------------------
  {
    const int rb   = w & 1;
    const int row0 = rb * 16;
    const int cg   = w >> 1;
    for (int cbi = 0; cbi < 32; ++cbi) {
      const int cb = cg + cbi * 4;
      const int n0 = cb * 16;
      v8f acc;
      #pragma unroll
      for (int i = 0; i < 8; ++i) acc[i] = 0.0f;

      #pragma unroll
      for (int kc4 = 0; kc4 < 4; ++kc4) {
        const int kc = kc4 * 32;
        // A: 16x32 f16 from LDS Q tile (contiguous: lanes0-15 K=0..15, lanes16-31 K=16..31)
        const _Float16* ap = Qs + (row0 + l16) * D + kc + lhi * 16;
        h8 alo = *reinterpret_cast<const h8*>(ap);
        h8 ahi = *reinterpret_cast<const h8*>(ap + 8);
        v16h a = __builtin_shufflevector(alo, ahi,
            0,1,2,3,4,5,6,7,8,9,10,11,12,13,14,15);
        // B: 32x16 f16 = K^T chunk; lane n=l16, K = lhi*16 + h (contiguous 16)
        const int n = n0 + l16;
        const float4* kp = reinterpret_cast<const float4*>(
            K + ((size_t)b * L + n) * D + kc + lhi * 16);
        float4 f0 = kp[0], f1 = kp[1], f2 = kp[2], f3 = kp[3];
        v16h bm;
        bm[0]=(_Float16)f0.x; bm[1]=(_Float16)f0.y; bm[2]=(_Float16)f0.z; bm[3]=(_Float16)f0.w;
        bm[4]=(_Float16)f1.x; bm[5]=(_Float16)f1.y; bm[6]=(_Float16)f1.z; bm[7]=(_Float16)f1.w;
        bm[8]=(_Float16)f2.x; bm[9]=(_Float16)f2.y; bm[10]=(_Float16)f2.z; bm[11]=(_Float16)f2.w;
        bm[12]=(_Float16)f3.x; bm[13]=(_Float16)f3.y; bm[14]=(_Float16)f3.z; bm[15]=(_Float16)f3.w;

        acc = wmma_f16(a, bm, acc);
      }
      // C/D layout: lane n=l16; vgpr r -> row row0 + lhi*8 + r
      const int col   = n0 + l16;
      const int rbase = row0 + lhi * 8;
      #pragma unroll
      for (int r2 = 0; r2 < 8; ++r2)
        S[(size_t)(rbase + r2) * L + col] = acc[r2];
    }
  }
  __syncthreads();

  // ------------------------------------------------------------------
  // Phase 2: softmax per row (wave w owns rows 4w..4w+3), lane-interleaved
  //   sweep1: scale + mask -> store back, row max
  //   sweep2: exp(s - max)  -> store back, row sum
  //   sweep3: normalize     -> NT store attn, pack f16 P in place
  // ------------------------------------------------------------------
  for (int rr = 0; rr < 4; ++rr) {
    const int r  = w * 4 + rr;
    const int gq = q0 + r;
    const uint8_t* mrow = MASK + ((size_t)b * L + gq) * L;
    float* srow = S + (size_t)r * L;

    float mx = -__builtin_inff();
    for (int i = 0; i < 64; ++i) {
      int c = i * 32 + lane;
      float s = srow[c] * INV_TEMPER;
      if (__builtin_nontemporal_load(mrow + c)) s = -__builtin_inff();
      srow[c] = s;
      mx = fmaxf(mx, s);
    }
    #pragma unroll
    for (int off = 16; off; off >>= 1) mx = fmaxf(mx, __shfl_xor(mx, off, 32));

    float sum = 0.0f;
    for (int i = 0; i < 64; ++i) {
      int c = i * 32 + lane;
      float e = __expf(srow[c] - mx);
      srow[c] = e;
      sum += e;
    }
    #pragma unroll
    for (int off = 16; off; off >>= 1) sum += __shfl_xor(sum, off, 32);
    const float inv = 1.0f / sum;

    float* arow = ATTN + ((size_t)b * L + gq) * L;
    _Float16* prow = Ph + (size_t)r * (2 * L);   // P row overlays first half of S row
    for (int i = 0; i < 64; ++i) {
      int c = i * 32 + lane;
      float p = srow[c] * inv;                    // ds_load before ds_store: wave-ordered, safe
      __builtin_nontemporal_store(p, arow + c);   // stream 268MB past L2
      prow[c] = (_Float16)p;
    }
  }
  __syncthreads();

  // ------------------------------------------------------------------
  // Phase 3: O = P V.  16 output tiles (2 rb x 8 cb); wave w owns 2 tiles.
  //   Each 32-key chunk of V is staged transposed->f16 in LDS, shared by all.
  // ------------------------------------------------------------------
  const int t0 = w * 2, t1 = w * 2 + 1;
  const int rb0 = t0 >> 3, cb0 = t0 & 7;
  const int rb1 = t1 >> 3, cb1 = t1 & 7;
  v8f o0, o1;
  #pragma unroll
  for (int i = 0; i < 8; ++i) { o0[i] = 0.0f; o1[i] = 0.0f; }

  for (int kk = 0; kk < L; kk += 32) {
    // stage V chunk transposed: Vt[d][k_local], f16
    #pragma unroll
    for (int i = 0; i < 16; ++i) {
      int e = i * 256 + tid;
      int rl = e >> 7, d = e & 127;
      Vt[d * 32 + rl] = (_Float16)V[((size_t)b * L + kk + rl) * D + d];
    }
    __syncthreads();

    auto loadA = [&](int rbx) -> v16h {
      const _Float16* pp = Ph + (size_t)(rbx * 16 + l16) * (2 * L) + kk + lhi * 16;
      h8 lo = *reinterpret_cast<const h8*>(pp);
      h8 hi = *reinterpret_cast<const h8*>(pp + 8);
      return __builtin_shufflevector(lo, hi,
          0,1,2,3,4,5,6,7,8,9,10,11,12,13,14,15);
    };
    auto loadB = [&](int cbx) -> v16h {
      const _Float16* vp = Vt + (cbx * 16 + l16) * 32 + lhi * 16;
      h8 lo = *reinterpret_cast<const h8*>(vp);
      h8 hi = *reinterpret_cast<const h8*>(vp + 8);
      return __builtin_shufflevector(lo, hi,
          0,1,2,3,4,5,6,7,8,9,10,11,12,13,14,15);
    };

    o0 = wmma_f16(loadA(rb0), loadB(cb0), o0);
    o1 = wmma_f16(loadA(rb1), loadB(cb1), o1);
    __syncthreads();
  }

  auto storeO = [&](v8f o, int rbx, int cbx) {
    const int n     = cbx * 16 + l16;
    const int rbase = rbx * 16 + lhi * 8;
    #pragma unroll
    for (int r2 = 0; r2 < 8; ++r2)
      __builtin_nontemporal_store(
          o[r2], OUT + ((size_t)b * L + q0 + rbase + r2) * D + n);
  };
  storeO(o0, rb0, cb0);
  storeO(o1, rb1, cb1);
}

extern "C" void kernel_launch(void* const* d_in, const int* in_sizes, int n_in,
                              void* d_out, int out_size, void* d_ws, size_t ws_size,
                              hipStream_t stream) {
  (void)in_sizes; (void)n_in; (void)out_size; (void)d_ws; (void)ws_size;
  const float*   q    = (const float*)d_in[0];
  const float*   k    = (const float*)d_in[1];
  const float*   v    = (const float*)d_in[2];
  const uint8_t* mask = (const uint8_t*)d_in[3];   // jax bool -> 1 byte/elem

  float* out  = (float*)d_out;                      // [16,2048,128]
  float* attn = out + (size_t)16 * L * D;           // [16,2048,2048]

  hipFuncSetAttribute((const void*)sdpa_attn_kernel,
                      hipFuncAttributeMaxDynamicSharedMemorySize,
                      (int)SMEM_SZ);

  dim3 grid(L / MTILE, 16);   // 64 q-tiles x 16 batches
  dim3 block(256);            // 8 wave32
  sdpa_attn_kernel<<<grid, block, SMEM_SZ, stream>>>(q, k, v, mask, out, attn);
}
